// Attention_21191368639408
// MI455X (gfx1250) — matrix operand order repaired and checked by reading, not compiled, not run on hardware
//
#include <hip/hip_runtime.h>
#include <hip/hip_bf16.h>
#include <stdint.h>

typedef __attribute__((ext_vector_type(16))) __bf16 v16bf;
typedef __attribute__((ext_vector_type(8)))  float  v8f;
typedef unsigned int u32x4 __attribute__((ext_vector_type(4)));
typedef int          i32x8 __attribute__((ext_vector_type(8)));
typedef int          i32x4 __attribute__((ext_vector_type(4)));

#define BATCH 8
#define NTOK  1024
#define DIM   768
#define HEADS 12
#define HD    64
#define BH    (BATCH * HEADS)
#define KT24  24                   // 768 / 32 K-chunks for the big GEMMs

union ABu { v16bf v; uint32_t u[8]; uint4 q[2]; };

static __device__ __forceinline__ v8f wmma_bf16(v16bf a, v16bf b, v8f c) {
  return __builtin_amdgcn_wmma_f32_16x16x32_bf16(false, a, false, b, (short)0, c,
                                                 false, false);
}

// TDM: contiguous 1-row 2D copy of nElem4B dwords from global to LDS.
// D# built per CDNA5 ISA (group0/group1 bit layout).
static __device__ __forceinline__ void tdm_load_1d(const void* gptr, void* lptr,
                                                   int nElem4B) {
  uint64_t ga = (uint64_t)(uintptr_t)gptr;
  uint32_t la = (uint32_t)(uintptr_t)lptr;        // LDS aperture low bits
  u32x4 g0;
  g0[0] = 1u;                                     // count=1 (valid user D#)
  g0[1] = la;                                     // lds_addr
  g0[2] = (uint32_t)ga;                           // global_addr[31:0]
  g0[3] = (uint32_t)((ga >> 32) & 0x01FFFFFFu) | (2u << 30);  // addr[56:32]|type=2
  i32x8 g1;
  g1[0] = (2 << 16);                              // workgroup_mask=0, data_size=4B
  g1[1] = (int)((nElem4B & 0xFFFF) << 16);        // tensor_dim0[15:0] @ bits 63:48
  g1[2] = (int)(((unsigned)nElem4B >> 16) & 0xFFFF) | (1 << 16); // dim0 hi, dim1=1 lo
  g1[3] = (int)((nElem4B & 0xFFFF) << 16);        // dim1 hi=0, tile_dim0
  g1[4] = 1;                                      // tile_dim1=1, tile_dim2=0
  g1[5] = nElem4B;                                // tensor_dim0_stride lo32
  g1[6] = 0;                                      // stride hi, dim1_stride lo
  g1[7] = 0;
  i32x4 z4 = {0, 0, 0, 0};                        // groups 2/3: tile_dim3/4 unused
#if defined(__clang_major__) && __clang_major__ >= 23
  i32x8 z8 = {0, 0, 0, 0, 0, 0, 0, 0};
  __builtin_amdgcn_tensor_load_to_lds(g0, g1, z4, z4, z8, 0);
#else
  __builtin_amdgcn_tensor_load_to_lds(g0, g1, z4, z4, 0);
#endif
}

// Element offset of (row, c) inside an A-operand-swizzled bf16 buffer.
// Tile = 512 elems (32 lanes x 16 bf16), row-tile-major. kTiles = K/32.
static __device__ __forceinline__ size_t aswz_off(int row, int c, int kTiles) {
  int c32 = c & 31;
  int ln  = ((c32 >> 3) & 1) * 16 + (row & 15);
  int r   = ((c32 >> 4) << 2) | ((c32 & 7) >> 1);
  return (((size_t)(row >> 4) * kTiles + (c >> 5)) << 9) + ln * 16 + r * 2 + (c32 & 1);
}
// Element offset of (col, c) inside a B-operand-swizzled bf16 buffer,
// K-chunk-major tile order: tile_id = (c>>5)*nColTiles + (col>>4).
static __device__ __forceinline__ size_t bswz_off(int col, int c, int nColTiles) {
  int c32 = c & 31;
  int ln  = ((c32 >> 3) & 1) * 16 + (col & 15);
  int r   = ((c32 >> 4) << 2) | ((c32 & 7) >> 1);
  return (((size_t)(c >> 5) * nColTiles + (col >> 4)) << 9) + ln * 16 + r * 2 + (c32 & 1);
}

// ---------------------------------------------------------------------------
// Pre-pass: f32 (rows x 768 row-major) -> bf16 A-operand swizzle
// ---------------------------------------------------------------------------
__global__ __launch_bounds__(256) void swizzleA_f32_kernel(
    const float* __restrict__ src, uint32_t* __restrict__ dst, int totalPairs)
{
  int t = blockIdx.x * 256 + threadIdx.x;
  if (t >= totalPairs) return;
  int tile = t >> 8, lane = (t >> 3) & 31, r = t & 7;
  int mt = tile / KT24, kc = tile - mt * KT24;
  int m = mt * 16 + (lane & 15);
  int k = kc * 32 + ((r < 4) ? 0 : 16) + (lane >> 4) * 8 + (r & 3) * 2;
  const float* s = src + (size_t)m * DIM + k;
  union { __bf16 h[2]; uint32_t w; } pk;
  pk.h[0] = (__bf16)s[0]; pk.h[1] = (__bf16)s[1];
  dst[t] = pk.w;
}

// ---------------------------------------------------------------------------
// Pre-pass: f32 weight (ncols x 768 row-major) -> bf16 B-operand swizzle,
// K-chunk-major so the per-(block,kc) panel is 8 KB contiguous for the TDM.
// ---------------------------------------------------------------------------
__global__ __launch_bounds__(256) void swizzleB_f32_kernel(
    const float* __restrict__ src, uint32_t* __restrict__ dst,
    int nColTiles, int totalPairs)
{
  int t = blockIdx.x * 256 + threadIdx.x;
  if (t >= totalPairs) return;
  int tile = t >> 8, lane = (t >> 3) & 31, r = t & 7;
  int kc = tile / nColTiles, ct = tile - kc * nColTiles;
  int col = ct * 16 + (lane & 15);
  int c = kc * 32 + ((r < 4) ? 0 : 16) + (lane >> 4) * 8 + (r & 3) * 2;
  const float* s = src + (size_t)col * DIM + c;
  union { __bf16 h[2]; uint32_t w; } pk;
  pk.h[0] = (__bf16)s[0]; pk.h[1] = (__bf16)s[1];
  dst[t] = pk.w;
}

// ---------------------------------------------------------------------------
// Kernel 1: qkv = x @ qkv_w^T + b -> swizzled Q(A) / K(B) / V(B over key dim)
// block = 256 (8 waves), wave tile = 16 x 128. B panel TDM-staged in LDS,
// double buffered: DMA of chunk kc+1 overlaps the 8 WMMAs of chunk kc.
// ---------------------------------------------------------------------------
__global__ __launch_bounds__(256) void qkv_gemm_kernel(
    const __bf16* __restrict__ xsw, const __bf16* __restrict__ wsw,
    const float* __restrict__ bias,
    __bf16* __restrict__ qsw, __bf16* __restrict__ ksw, __bf16* __restrict__ vsw)
{
  __shared__ __align__(16) unsigned char s_b[2][8192];

  const int tid  = threadIdx.x;
  const int wave = tid >> 5, lane = tid & 31;
  const int lh = lane & 15, hf = lane >> 4;
  const int mt  = blockIdx.y * 8 + wave;    // 16-row tile (0..511)
  const int nt0 = blockIdx.x * 8;           // 16-col tile base (0..143)

  if (wave == 0) {
    tdm_load_1d(wsw + (((size_t)0 * 144 + nt0) << 9), &s_b[0][0], 2048);
    __builtin_amdgcn_s_wait_tensorcnt(0);
  }
  __syncthreads();

  v8f acc[8] = {};
  for (int kc = 0; kc < KT24; ++kc) {
    if (wave == 0 && kc + 1 < KT24)
      tdm_load_1d(wsw + (((size_t)(kc + 1) * 144 + nt0) << 9),
                  &s_b[(kc + 1) & 1][0], 2048);
    ABu a;
    const uint4* ap = (const uint4*)(xsw + (((size_t)mt * KT24 + kc) << 9)) + lane * 2;
    a.q[0] = ap[0]; a.q[1] = ap[1];
    ABu b[8];
#pragma unroll
    for (int nt = 0; nt < 8; ++nt) {
      const uint4* bp = (const uint4*)(&s_b[kc & 1][nt * 1024]) + lane * 2;
      b[nt].q[0] = bp[0]; b[nt].q[1] = bp[1];
    }
#pragma unroll
    for (int nt = 0; nt < 8; ++nt)
      acc[nt] = wmma_bf16(a.v, b[nt].v, acc[nt]);
    if (wave == 0 && kc + 1 < KT24)
      __builtin_amdgcn_s_wait_tensorcnt(0);
    __syncthreads();
  }

#pragma unroll
  for (int nt = 0; nt < 8; ++nt) {
    int ncol = (nt0 + nt) * 16 + lh;
    float bv = bias[ncol];
    int sec = ncol / DIM;                   // 0:q 1:k 2:v
    int jj  = ncol - sec * DIM;
    int head = jj >> 6, ch = jj & 63;
#pragma unroll
    for (int r = 0; r < 8; ++r) {
      int mg = mt * 16 + r + hf * 8;        // C layout row
      int bi = mg >> 10, ni = mg & 1023;
      int bh = bi * HEADS + head;
      __bf16 val = (__bf16)(acc[nt][r] + bv);
      if (sec == 0)      qsw[(size_t)bh * 65536 + aswz_off(ni, ch, 2)]  = val;
      else if (sec == 1) ksw[(size_t)bh * 65536 + bswz_off(ni, ch, 64)] = val;
      else               vsw[(size_t)bh * 65536 + bswz_off(ch, ni, 4)]  = val;
    }
  }
}

static __device__ __forceinline__ float q_elem(const __bf16* qsw, int bh, int n, int c) {
  return (float)qsw[(size_t)bh * 65536 + aswz_off(n, c, 2)];
}

// ---------------------------------------------------------------------------
// Kernel 2: flash attention with decomposed rel-pos bias
// grid = (N/64, B*HEADS), block = 128 (4 waves, 16 query rows each)
// K buffer: B-swizzle, K-chunk-major over keys -> tile = (c>>5)*64 + (key>>4)
// V buffer: B-swizzle over key dim       -> tile = (key>>5)*4 + (ch>>4)
// ---------------------------------------------------------------------------
__global__ __launch_bounds__(128) void attn_kernel(
    const __bf16* __restrict__ qsw, const __bf16* __restrict__ ksw,
    const __bf16* __restrict__ vsw,
    const float* __restrict__ rel_h, const float* __restrict__ rel_w,
    __bf16* __restrict__ aosw)
{
  __shared__ float s_relh[64][32];
  __shared__ float s_relw[64][32];
  __shared__ __align__(16) unsigned char s_p[4][2048];  // per-wave P in A-swizzle

  const int tid = threadIdx.x;
  const int bh  = blockIdx.y;
  const int n0  = blockIdx.x * 64;
  const int bidx = bh / HEADS, head = bh - bidx * HEADS;

  for (int i = 0; i < 32; ++i) {
    int e   = tid + i * 128;
    int isW = e >> 11;
    int ee  = e & 2047;
    int m = ee >> 5, kk = ee & 31;
    int nq = n0 + m;
    const float* rr = isW ? (rel_w + (size_t)((nq & 31) - kk + 31) * HD)
                          : (rel_h + (size_t)((nq >> 5) - kk + 31) * HD);
    float s = 0.f;
#pragma unroll 8
    for (int c = 0; c < HD; ++c) s += q_elem(qsw, bh, nq, c) * rr[c];
    if (isW) s_relw[m][kk] = s; else s_relh[m][kk] = s;
  }
  __syncthreads();

  const int wave = tid >> 5, lane = tid & 31;
  const int lh = lane & 15, hf = lane >> 4;

  ABu aq[2];
#pragma unroll
  for (int ck = 0; ck < 2; ++ck) {
    const uint4* p = (const uint4*)(qsw + (size_t)bh * 65536 +
                     (((size_t)((n0 >> 4) + wave) * 2 + ck) << 9)) + lane * 2;
    aq[ck].q[0] = p[0]; aq[ck].q[1] = p[1];
  }

  float mrow[8], lrow[8];
#pragma unroll
  for (int r = 0; r < 8; ++r) { mrow[r] = -3.0e38f; lrow[r] = 0.f; }
  v8f o[4] = {};

  for (int kt = 0; kt < 16; ++kt) {
    if (kt + 1 < 16) {   // gfx1250 global_prefetch for next key/value tiles
      __builtin_prefetch(ksw + (size_t)bh * 65536 + (((size_t)(kt + 1) * 4) << 9), 0, 1);
      __builtin_prefetch(vsw + (size_t)bh * 65536 + (((size_t)(kt + 1) * 8) << 9), 0, 1);
    }

    // ---- K B-tiles: tile id = ck*64 + (key>>4) ----
    ABu bk[4][2];
#pragma unroll
    for (int nt = 0; nt < 4; ++nt)
#pragma unroll
      for (int ck = 0; ck < 2; ++ck) {
        const uint4* p = (const uint4*)(ksw + (size_t)bh * 65536 +
                         (((size_t)ck * 64 + kt * 4 + nt) << 9)) + lane * 2;
        bk[nt][ck].q[0] = p[0]; bk[nt][ck].q[1] = p[1];
      }
    v8f s[4];
#pragma unroll
    for (int nt = 0; nt < 4; ++nt) {
      v8f z = {};
      v8f t = wmma_bf16(aq[0].v, bk[nt][0].v, z);
      s[nt] = wmma_bf16(aq[1].v, bk[nt][1].v, t);
    }

    // ---- V B-tiles loaded early: latency hides under softmax VALU ----
    ABu bv[4][2];
#pragma unroll
    for (int nt = 0; nt < 4; ++nt)
#pragma unroll
      for (int ck = 0; ck < 2; ++ck) {
        const uint4* pv = (const uint4*)(vsw + (size_t)bh * 65536 +
                          (((size_t)(kt * 2 + ck) * 4 + nt) << 9)) + lane * 2;
        bv[nt][ck].q[0] = pv[0]; bv[nt][ck].q[1] = pv[1];
      }

    // ---- scale + bias, online softmax (C layout: row = r + hf*8) ----
    float p[4][8], rmax[8];
#pragma unroll
    for (int r = 0; r < 8; ++r) rmax[r] = -3.0e38f;
#pragma unroll
    for (int nt = 0; nt < 4; ++nt) {
      int nk = kt * 64 + nt * 16 + lh;
      int ky = nk >> 5, kx = nk & 31;
#pragma unroll
      for (int r = 0; r < 8; ++r) {
        int qm = wave * 16 + r + hf * 8;
        float v = s[nt][r] * 0.125f + s_relh[qm][ky] + s_relw[qm][kx];
        p[nt][r] = v;
        rmax[r] = fmaxf(rmax[r], v);
      }
    }
#pragma unroll
    for (int r = 0; r < 8; ++r)
#pragma unroll
      for (int off = 8; off > 0; off >>= 1)
        rmax[r] = fmaxf(rmax[r], __shfl_xor(rmax[r], off, 32));

    float alpha[8], rsum[8];
#pragma unroll
    for (int r = 0; r < 8; ++r) {
      float mn = fmaxf(mrow[r], rmax[r]);
      alpha[r] = __expf(mrow[r] - mn);
      mrow[r] = mn;
      rsum[r] = 0.f;
    }
#pragma unroll
    for (int nt = 0; nt < 4; ++nt)
#pragma unroll
      for (int r = 0; r < 8; ++r) {
        float e = __expf(p[nt][r] - mrow[r]);
        p[nt][r] = e;
        rsum[r] += e;
      }
#pragma unroll
    for (int r = 0; r < 8; ++r) {
#pragma unroll
      for (int off = 8; off > 0; off >>= 1)
        rsum[r] += __shfl_xor(rsum[r], off, 32);
      lrow[r] = lrow[r] * alpha[r] + rsum[r];
    }
#pragma unroll
    for (int nt = 0; nt < 4; ++nt)
#pragma unroll
      for (int r = 0; r < 8; ++r)
        o[nt][r] *= alpha[r];

    // ---- P: C layout -> A-operand swizzle in wave-private LDS ----
#pragma unroll
    for (int nt = 0; nt < 4; ++nt)
#pragma unroll
      for (int r = 0; r < 8; ++r) {
        int m_loc = r + hf * 8;
        int c = nt * 16 + lh;
        int c32 = c & 31;
        int ln = ((c32 >> 3) & 1) * 16 + m_loc;
        int rr = ((c32 >> 4) << 2) | ((c32 & 7) >> 1);
        int off = (c >> 5) * 1024 + ln * 32 + rr * 4 + (c32 & 1) * 2;
        *(__bf16*)(&s_p[wave][off]) = (__bf16)p[nt][r];
      }
    ABu apk[2];
#pragma unroll
    for (int ck = 0; ck < 2; ++ck) {
      const uint4* pp = (const uint4*)(&s_p[wave][ck * 1024 + lane * 32]);
      apk[ck].q[0] = pp[0]; apk[ck].q[1] = pp[1];
    }

#pragma unroll
    for (int nt = 0; nt < 4; ++nt) {
      o[nt] = wmma_bf16(apk[0].v, bv[nt][0].v, o[nt]);
      o[nt] = wmma_bf16(apk[1].v, bv[nt][1].v, o[nt]);
    }
  }

  // ---- normalize, store into A-operand swizzle for proj GEMM ----
#pragma unroll
  for (int nt = 0; nt < 4; ++nt)
#pragma unroll
    for (int r = 0; r < 8; ++r) {
      int nrow = n0 + wave * 16 + r + hf * 8;
      int mg = bidx * NTOK + nrow;          // row in 8192
      int chfull = head * HD + nt * 16 + lh;
      float val = o[nt][r] / lrow[r];
      aosw[aswz_off(mg, chfull, KT24)] = (__bf16)val;
    }
}

// ---------------------------------------------------------------------------
// Kernel 3: out = attn_out @ proj_w^T + proj_b (f32 output), TDM-staged B
// ---------------------------------------------------------------------------
__global__ __launch_bounds__(256) void proj_gemm_kernel(
    const __bf16* __restrict__ aosw, const __bf16* __restrict__ wsw,
    const float* __restrict__ bias, float* __restrict__ out)
{
  __shared__ __align__(16) unsigned char s_b[2][8192];

  const int tid  = threadIdx.x;
  const int wave = tid >> 5, lane = tid & 31;
  const int lh = lane & 15, hf = lane >> 4;
  const int mt  = blockIdx.y * 8 + wave;
  const int nt0 = blockIdx.x * 8;           // 16-col tile base (0..47)

  if (wave == 0) {
    tdm_load_1d(wsw + (((size_t)0 * 48 + nt0) << 9), &s_b[0][0], 2048);
    __builtin_amdgcn_s_wait_tensorcnt(0);
  }
  __syncthreads();

  v8f acc[8] = {};
  for (int kc = 0; kc < KT24; ++kc) {
    if (wave == 0 && kc + 1 < KT24)
      tdm_load_1d(wsw + (((size_t)(kc + 1) * 48 + nt0) << 9),
                  &s_b[(kc + 1) & 1][0], 2048);
    ABu a;
    const uint4* ap = (const uint4*)(aosw + (((size_t)mt * KT24 + kc) << 9)) + lane * 2;
    a.q[0] = ap[0]; a.q[1] = ap[1];
    ABu b[8];
#pragma unroll
    for (int nt = 0; nt < 8; ++nt) {
      const uint4* bp = (const uint4*)(&s_b[kc & 1][nt * 1024]) + lane * 2;
      b[nt].q[0] = bp[0]; b[nt].q[1] = bp[1];
    }
#pragma unroll
    for (int nt = 0; nt < 8; ++nt)
      acc[nt] = wmma_bf16(a.v, b[nt].v, acc[nt]);
    if (wave == 0 && kc + 1 < KT24)
      __builtin_amdgcn_s_wait_tensorcnt(0);
    __syncthreads();
  }

#pragma unroll
  for (int nt = 0; nt < 8; ++nt) {
    int ncol = (nt0 + nt) * 16 + lh;
    float bv = bias[ncol];
#pragma unroll
    for (int r = 0; r < 8; ++r) {
      int mg = mt * 16 + r + hf * 8;
      out[(size_t)mg * DIM + ncol] = acc[nt][r] + bv;
    }
  }
}

// ---------------------------------------------------------------------------
extern "C" void kernel_launch(void* const* d_in, const int* in_sizes, int n_in,
                              void* d_out, int out_size, void* d_ws, size_t ws_size,
                              hipStream_t stream) {
  const float* x      = (const float*)d_in[0];
  const float* qkv_w  = (const float*)d_in[1];
  const float* qkv_b  = (const float*)d_in[2];
  const float* proj_w = (const float*)d_in[3];
  const float* proj_b = (const float*)d_in[4];
  const float* rel_h  = (const float*)d_in[5];
  const float* rel_w  = (const float*)d_in[6];
  float* out = (float*)d_out;

  const size_t xswN   = (size_t)512 * KT24 * 512;   // 6.29M elems
  const size_t wqkvN  = (size_t)144 * KT24 * 512;   // 1.77M
  const size_t wprojN = (size_t)48  * KT24 * 512;   // 0.59M
  const size_t qkvN   = (size_t)BH * 65536;         // 6.29M each

  __bf16* xsw   = (__bf16*)d_ws;
  __bf16* wqkv  = xsw + xswN;
  __bf16* wproj = wqkv + wqkvN;
  __bf16* qsw   = wproj + wprojN;
  __bf16* ksw   = qsw + qkvN;
  __bf16* vsw   = ksw + qkvN;
  __bf16* aosw  = vsw + qkvN;

  {
    int pairsX = 8192 * DIM / 2;
    swizzleA_f32_kernel<<<(pairsX + 255) / 256, 256, 0, stream>>>(
        x, (uint32_t*)xsw, pairsX);
    int pairsQW = 3 * DIM * DIM / 2;
    swizzleB_f32_kernel<<<(pairsQW + 255) / 256, 256, 0, stream>>>(
        qkv_w, (uint32_t*)wqkv, 144, pairsQW);
    int pairsPW = DIM * DIM / 2;
    swizzleB_f32_kernel<<<(pairsPW + 255) / 256, 256, 0, stream>>>(
        proj_w, (uint32_t*)wproj, 48, pairsPW);
  }

  qkv_gemm_kernel<<<dim3(2304 / 128, 8192 / 128), 256, 0, stream>>>(
      xsw, wqkv, qkv_b, qsw, ksw, vsw);
  attn_kernel<<<dim3(NTOK / 64, BH), 128, 0, stream>>>(
      qsw, ksw, vsw, rel_h, rel_w, aosw);
  proj_gemm_kernel<<<dim3(DIM / 128, 8192 / 128), 256, 0, stream>>>(
      aosw, wproj, proj_b, out);
}
